// DeformableConv2d_58076547776726
// MI455X (gfx1250) — matrix operand order repaired and checked by reading, compiled, not run on hardware
//
#include <hip/hip_runtime.h>
#include <math.h>

typedef __attribute__((ext_vector_type(16))) __bf16 v16bf;
typedef __attribute__((ext_vector_type(8)))  float v8f;
typedef __attribute__((ext_vector_type(4)))  unsigned int v4u;

#define BN 8
#define CI 64
#define HH 128
#define WW 128
#define OO 64
#define KTOT 576   // C * 9
#define NCHUNK 18  // 576 / 32

// workspace layout (bytes)
#define WS_A_OFF    0u          // 32x576 bf16 = 36864
#define WS_B_OFF    36864u      // 64x576 bf16 = 73728
#define WS_OFFSET   110592u     // 8*18*128*128 f32 = 9437184
#define WS_MASK     9547776u    // 8*9*128*128 f32 = 4718592

// conv kernel dynamic LDS partition (halves)
#define XS_HALVES    (CI * 3 * 66)      // 12672 (25344 B)
#define PATCH_STRIDE 584                // 576 + 8 pad
#define PATCH_HALVES (64 * PATCH_STRIDE)
#define CONV_LDS_BYTES ((XS_HALVES + PATCH_HALVES) * 2)

__device__ __forceinline__ unsigned short f2bf(float f) {
    unsigned int u = __float_as_uint(f);
    u += 0x7FFFu + ((u >> 16) & 1u);
    return (unsigned short)(u >> 16);
}

union BFrag {
    v16bf v;
    unsigned short u[16];
    v4u q[2];
};

// ---------------- Kernel 0: pack weights to bf16 ----------------
__global__ void pack_weights_kernel(const float* __restrict__ offw,
                                    const float* __restrict__ modw,
                                    const float* __restrict__ w,
                                    unsigned short* __restrict__ wsA,
                                    unsigned short* __restrict__ wsB) {
    int i = blockIdx.x * 256 + threadIdx.x;
    if (i < 32 * KTOT) {
        int o = i / KTOT, k = i - o * KTOT;
        float v = 0.f;
        if (o < 18)       v = offw[o * KTOT + k];
        else if (o < 27)  v = modw[(o - 18) * KTOT + k];
        wsA[i] = f2bf(v);
    }
    if (i < OO * KTOT) {
        wsB[i] = f2bf(w[i]);
    }
}

// ---------------- Kernel 1: fused offset+modulator conv (WMMA) ----------------
__global__ __launch_bounds__(128) void conv_offmod_kernel(
        const float* __restrict__ xg,
        const float* __restrict__ offb,
        const float* __restrict__ modb,
        const unsigned short* __restrict__ wsA,
        float* __restrict__ off,
        float* __restrict__ mask) {
    extern __shared__ unsigned short smem[];
    unsigned short* xs    = smem;               // [64ch][3row][66col] halo
    unsigned short* patch = smem + XS_HALVES;   // [64 col][584] im2col tile

    int bi = blockIdx.x;
    int xseg = bi & 1;
    int y = (bi >> 1) & (HH - 1);
    int b = bi >> 8;
    int base_x = xseg * 64;
    int tid = threadIdx.x;

    // prefetch the (tiny, shared) bf16 weight matrix while we stage LDS
    for (int i = tid; i < (32 * KTOT * 2) / 128; i += 128)
        __builtin_prefetch((const char*)wsA + i * 128, 0, 3);

    // phase 1: stage x halo into LDS as bf16 (zero-padded borders)
    for (int idx = tid; idx < XS_HALVES; idx += 128) {
        int c = idx / 198; int r2 = idx - c * 198;
        int ry = r2 / 66;  int cx = r2 - ry * 66;
        int gy = y - 1 + ry, gx = base_x - 1 + cx;
        float v = 0.f;
        if (gy >= 0 && gy < HH && gx >= 0 && gx < WW)
            v = xg[((b * CI + c) * HH + gy) * WW + gx];
        xs[idx] = f2bf(v);
    }
    __syncthreads();

    // phase 2: expand halo -> im2col patch [col][c*9 + ky*3 + kx]
    // all tap offsets are compile-time immediates; c advances incrementally
    {
        int col = tid & 63;
        int chalf = tid >> 6;                       // this thread's c parity
        unsigned int rbase = (unsigned)(chalf * 198 + col);
        unsigned int wbase = (unsigned)(col * PATCH_STRIDE + chalf * 9);
#pragma unroll 1
        for (int c2 = 0; c2 < 32; ++c2) {
#pragma unroll
            for (int ky = 0; ky < 3; ++ky) {
#pragma unroll
                for (int kx = 0; kx < 3; ++kx) {
                    patch[wbase + ky * 3 + kx] = xs[rbase + ky * 66 + kx];
                }
            }
            rbase += 2 * 198;
            wbase += 2 * 9;
        }
    }
    __syncthreads();

    // phase 3: [32x576] x [576x16-per-wave] GEMM
    int lane = tid & 31, wv = tid >> 5;
    int coln = lane & 15, half = lane >> 4;
    int kbA = half ? 8 : 0;
    int kbB = half ? 8 : 0;
    int rowa = lane & 15;

    v8f acc0 = {}; v8f acc1 = {};
#pragma unroll
    for (int t = 0; t < NCHUNK; ++t) {
        BFrag a0, a1, bb;
        const v4u* p0 = (const v4u*)(wsA + rowa * KTOT + 32 * t);
        const v4u* p1 = (const v4u*)(wsA + (rowa + 16) * KTOT + 32 * t);
        int s = kbA >> 3;
        a0.q[0] = p0[s]; a0.q[1] = p0[s + 2];
        a1.q[0] = p1[s]; a1.q[1] = p1[s + 2];
        const v4u* pb = (const v4u*)(patch + (wv * 16 + coln) * PATCH_STRIDE
                                     + 32 * t + kbB);
        bb.q[0] = pb[0]; bb.q[1] = pb[2];
        acc0 = __builtin_amdgcn_wmma_f32_16x16x32_bf16(false, a0.v, false, bb.v,
                                                       (short)0, acc0, false, false);
        acc1 = __builtin_amdgcn_wmma_f32_16x16x32_bf16(false, a1.v, false, bb.v,
                                                       (short)0, acc1, false, false);
    }

    int xpos = base_x + wv * 16 + coln;
#pragma unroll
    for (int r = 0; r < 8; ++r) {
        int o0 = r + 8 * half;        // rows 0..15: offset channels
        int o1 = 16 + o0;             // rows 16..31: offset(16,17)/mod(18..26)
        {
            float vv = acc0[r] + offb[o0];
            vv = fminf(32.f, fmaxf(-32.f, vv));
            off[((b * 18 + o0) * HH + y) * WW + xpos] = vv;
        }
        if (o1 < 18) {
            float vv = acc1[r] + offb[o1];
            vv = fminf(32.f, fmaxf(-32.f, vv));
            off[((b * 18 + o1) * HH + y) * WW + xpos] = vv;
        } else if (o1 < 27) {
            float vv = acc1[r] + modb[o1 - 18];
            mask[((b * 9 + (o1 - 18)) * HH + y) * WW + xpos] =
                2.f / (1.f + __expf(-vv));
        }
    }
}

// ---------------- Kernel 2: deformable gather + WMMA GEMM ----------------
__global__ __launch_bounds__(256) void deform_wmma_kernel(
        const float* __restrict__ xg,
        const float* __restrict__ off,
        const float* __restrict__ mask,
        const unsigned short* __restrict__ wsB,
        float* __restrict__ out) {
    __shared__ int s_y0[288], s_x0[288];
    __shared__ float s_w[288 * 4];
    __shared__ unsigned short s_val[32 * PATCH_STRIDE];

    int tid = threadIdx.x;
    int lane = tid & 31, wv = tid >> 5;
    int nbase = blockIdx.x * 32;
    int b = nbase >> 14;            // / (H*W)
    int rem = nbase & 16383;
    int y = rem >> 7;
    int xb = rem & 127;

    // prefetch this wave's A-matrix rows while the gather phases run
    {
        const char* wp = (const char*)(wsB + (wv >> 1) * 16 * KTOT);
        for (int i = lane; i < (16 * KTOT * 2) / 128; i += 32)
            __builtin_prefetch(wp + i * 128, 0, 3);
    }

    // phase 1: bilinear tap params per (k, column), mask folded into weights
    for (int i = tid; i < 288; i += 256) {
        int k = i >> 5, j = i & 31;
        int xgp = xb + j;
        float dy = off[((b * 18 + 2 * k    ) * HH + y) * WW + xgp];
        float dx = off[((b * 18 + 2 * k + 1) * HH + y) * WW + xgp];
        float m  = mask[((b * 9 + k) * HH + y) * WW + xgp];
        float py = (float)(y   + (k / 3) - 1) + dy;
        float px = (float)(xgp + (k % 3) - 1) + dx;
        float fy = floorf(py), fx = floorf(px);
        s_y0[i] = (int)fy; s_x0[i] = (int)fx;
        float wy = py - fy, wx = px - fx;
        s_w[4 * i + 0] = (1.f - wy) * (1.f - wx) * m;
        s_w[4 * i + 1] = (1.f - wy) * wx * m;
        s_w[4 * i + 2] = wy * (1.f - wx) * m;
        s_w[4 * i + 3] = wy * wx * m;
    }
    __syncthreads();

    // phase 2: gather val[576][32] tile into LDS bf16 (x is L2-resident)
    for (int idx = tid; idx < CI * 288; idx += 256) {
        int c = idx / 288; int r = idx - c * 288;
        int k = r >> 5, j = r & 31;
        int y0 = s_y0[r], x0 = s_x0[r];
        const float* xp = xg + ((b * CI + c) * HH) * WW;
        bool yv0 = (y0 >= 0)     & (y0 < HH);
        bool yv1 = (y0 + 1 >= 0) & (y0 + 1 < HH);
        bool xv0 = (x0 >= 0)     & (x0 < WW);
        bool xv1 = (x0 + 1 >= 0) & (x0 + 1 < WW);
        float v = 0.f;
        if (yv0 && xv0) v += s_w[4 * r + 0] * xp[y0 * WW + x0];
        if (yv0 && xv1) v += s_w[4 * r + 1] * xp[y0 * WW + x0 + 1];
        if (yv1 && xv0) v += s_w[4 * r + 2] * xp[(y0 + 1) * WW + x0];
        if (yv1 && xv1) v += s_w[4 * r + 3] * xp[(y0 + 1) * WW + x0 + 1];
        s_val[j * PATCH_STRIDE + c * 9 + k] = f2bf(v);
    }
    __syncthreads();

    // phase 3: [64x576] x [576x32] GEMM, 8 waves -> 4 o-tiles x 2 n-tiles
    int ot = wv >> 1, nt = wv & 1;
    int coln = lane & 15, half = lane >> 4;
    int kbA = half ? 8 : 0;
    int kbB = half ? 8 : 0;
    int rowa = ot * 16 + coln;

    v8f acc = {};
#pragma unroll
    for (int t = 0; t < NCHUNK; ++t) {
        BFrag a, bb;
        const v4u* pa = (const v4u*)(wsB + rowa * KTOT + 32 * t);
        int s = kbA >> 3;
        a.q[0] = pa[s]; a.q[1] = pa[s + 2];
        const v4u* pb = (const v4u*)(s_val + (nt * 16 + coln) * PATCH_STRIDE
                                     + 32 * t + kbB);
        bb.q[0] = pb[0]; bb.q[1] = pb[2];
        acc = __builtin_amdgcn_wmma_f32_16x16x32_bf16(false, a.v, false, bb.v,
                                                      (short)0, acc, false, false);
    }

    int xout = xb + nt * 16 + coln;
#pragma unroll
    for (int r = 0; r < 8; ++r) {
        int o = ot * 16 + r + 8 * half;
        out[((b * OO + o) * HH + y) * WW + xout] = acc[r];
    }
}

extern "C" void kernel_launch(void* const* d_in, const int* in_sizes, int n_in,
                              void* d_out, int out_size, void* d_ws, size_t ws_size,
                              hipStream_t stream) {
    const float* x    = (const float*)d_in[0];
    const float* offw = (const float*)d_in[1];
    const float* offb = (const float*)d_in[2];
    const float* modw = (const float*)d_in[3];
    const float* modb = (const float*)d_in[4];
    const float* w    = (const float*)d_in[5];
    float* out = (float*)d_out;

    char* ws = (char*)d_ws;
    unsigned short* wsA = (unsigned short*)(ws + WS_A_OFF);
    unsigned short* wsB = (unsigned short*)(ws + WS_B_OFF);
    float* off  = (float*)(ws + WS_OFFSET);
    float* mask = (float*)(ws + WS_MASK);

    pack_weights_kernel<<<(OO * KTOT + 255) / 256, 256, 0, stream>>>(
        offw, modw, w, wsA, wsB);

    // B * H * (W/64) blocks, 4 waves each -> 32x64 output tile per block
    conv_offmod_kernel<<<BN * HH * (WW / 64), 128, CONV_LDS_BYTES, stream>>>(
        x, offb, modb, wsA, off, mask);

    // (B*H*W)/32 blocks, 8 waves each -> 64x32 output tile per block
    deform_wmma_kernel<<<(BN * HH * WW) / 32, 256, 0, stream>>>(
        x, off, mask, wsB, out);
}
